// FeatureInterDependenciesModule_72765335929772
// MI455X (gfx1250) — compile-verified, matrix operand order checked by reading, not run on hardware
//
#include <hip/hip_runtime.h>
#include <hip/hip_bf16.h>
#include <math.h>

// ---------------------------------------------------------------------------
// Types for WMMA fragments (gfx1250, wave32)
// ---------------------------------------------------------------------------
typedef __bf16 bf16_t;
typedef bf16_t v16bf __attribute__((ext_vector_type(16)));
typedef float  v8f   __attribute__((ext_vector_type(8)));
typedef unsigned short u16;
typedef u16    u16x8 __attribute__((ext_vector_type(8)));

// Problem constants
#define NB    16
#define NC    256
#define IH    72
#define IW    88
#define HPOOL 36
#define WPOOL 44
#define MPIX  1584          // 36*44
#define KPAD  1600          // padded to multiple of 32 for bf16 WMMA K-steps
#define OUTH  145
#define OUTW  174
#define CR    85            // 256 // 3

#define MODE_GRAM  0        // f32 out = acc / MPIX
#define MODE_COPY  1        // bf16 out = acc
#define MODE_NS    2        // bf16 out = 0.5*(3I - acc)
#define MODE_FINAL 3        // f32 out = acc * scale[b]

__device__ __forceinline__ u16 f2bf(float f) {
  unsigned u = __float_as_uint(f);
  u += 0x7FFFu + ((u >> 16) & 1u);       // round-to-nearest-even
  return (u16)(u >> 16);
}

union FragU { v16bf v; u16x8 h[2]; };

// A/B fragment for v_wmma_f32_16x16x32_bf16 (wave32):
// lane l in [0,16): row/col = base + l, holds K = kBase+0..7 (h[0]) and
// kBase+16..23 (h[1]); lane 16+l holds K = kBase+8..15 / kBase+24..31.
__device__ __forceinline__ v16bf load_frag(const u16* mat, int ld, int rowBase,
                                           int kBase, int lane) {
  const int r    = rowBase + (lane & 15);
  const int half = lane >> 4;
  const u16* p = mat + (size_t)r * ld + kBase + half * 8;
  FragU f;
  f.h[0] = *(const u16x8*)p;
  f.h[1] = *(const u16x8*)(p + 16);
  return f.v;
}

// ---------------------------------------------------------------------------
// K1: 2x2 maxpool + per-(b,c) mean-centering + f32->bf16, zero-pad M to 1600.
// One block per (b,c) plane; 256 threads.
// ---------------------------------------------------------------------------
__global__ void pool_center_kernel(const float* __restrict__ x,
                                   u16* __restrict__ xc) {
  const int bc = blockIdx.x;            // 0..NB*NC-1
  const int t  = threadIdx.x;
  const float* p = x + (size_t)bc * (IH * IW);

  float pv[7];
  float lsum = 0.f;
  int cnt = 0;
  for (int pi = t; pi < MPIX; pi += 256) {
    const int py = pi / WPOOL, px = pi % WPOOL;
    const float* q = p + py * (2 * IW) + px * 2;
    float v = fmaxf(fmaxf(q[0], q[1]), fmaxf(q[IW], q[IW + 1]));
    pv[cnt++] = v;
    lsum += v;
  }

  __shared__ float red[256];
  red[t] = lsum;
  __syncthreads();
  for (int s = 128; s > 0; s >>= 1) {
    if (t < s) red[t] += red[t + s];
    __syncthreads();
  }
  const float mean = red[0] * (1.f / (float)MPIX);

  u16* row = xc + (size_t)bc * KPAD;
  cnt = 0;
  for (int pi = t; pi < MPIX; pi += 256) row[pi] = f2bf(pv[cnt++] - mean);
  if (t < KPAD - MPIX) row[MPIX + t] = 0;   // zero pad -> no Gram contribution
}

// ---------------------------------------------------------------------------
// K2/K5..: unified bf16 WMMA matmul, D = A * B for 256x256 output per batch.
// Both operands symmetric (or B=A^T for the Gram), so B columns are fetched
// as K-contiguous rows: identical fragment loads for A and B.
// One wave (32 threads) computes a 32x32 tile: 2x2 fragments, 4 wmma / K-step.
// grid = (8, 8, NB), block = 32.
// ---------------------------------------------------------------------------
__global__ void mm256_bf16_kernel(const u16* __restrict__ A,
                                  const u16* __restrict__ B,
                                  int lda, int ldb, int K,
                                  u16* __restrict__ outBf,
                                  float* __restrict__ outF,
                                  int mode,
                                  const float* __restrict__ scale) {
  const int lane = threadIdx.x;          // 0..31, wave32
  const int ti   = blockIdx.x * 32;
  const int tj   = blockIdx.y * 32;
  const int b    = blockIdx.z;

  const u16* Ab = A + (size_t)b * 256 * lda;
  const u16* Bb = B + (size_t)b * 256 * ldb;

  v8f acc[2][2];
  for (int i = 0; i < 2; ++i)
    for (int j = 0; j < 2; ++j)
      for (int r = 0; r < 8; ++r) acc[i][j][r] = 0.f;

  for (int kb = 0; kb < K; kb += 32) {
    if (kb + 32 < K) {   // hint next K-slab toward the caches
      __builtin_prefetch(Ab + (size_t)(ti + (lane & 15)) * lda + kb + 32, 0, 1);
      __builtin_prefetch(Bb + (size_t)(tj + (lane & 15)) * ldb + kb + 32, 0, 1);
    }
    v16bf a0 = load_frag(Ab, lda, ti,      kb, lane);
    v16bf a1 = load_frag(Ab, lda, ti + 16, kb, lane);
    v16bf b0 = load_frag(Bb, ldb, tj,      kb, lane);
    v16bf b1 = load_frag(Bb, ldb, tj + 16, kb, lane);

    acc[0][0] = __builtin_amdgcn_wmma_f32_16x16x32_bf16(
        false, a0, false, b0, (short)0, acc[0][0], false, false);
    acc[0][1] = __builtin_amdgcn_wmma_f32_16x16x32_bf16(
        false, a0, false, b1, (short)0, acc[0][1], false, false);
    acc[1][0] = __builtin_amdgcn_wmma_f32_16x16x32_bf16(
        false, a1, false, b0, (short)0, acc[1][0], false, false);
    acc[1][1] = __builtin_amdgcn_wmma_f32_16x16x32_bf16(
        false, a1, false, b1, (short)0, acc[1][1], false, false);
  }

  // C/D layout: lane<16 -> rows 0..7 in v[0..7], col=lane; lane>=16 -> rows 8..15.
  const int half = lane >> 4;
  const int nlo  = lane & 15;
  const size_t ob = (size_t)b * 256 * 256;
  const float sc = (mode == MODE_FINAL) ? scale[b] : 0.f;

  for (int i = 0; i < 2; ++i)
    for (int j = 0; j < 2; ++j)
      for (int r = 0; r < 8; ++r) {
        const int row = ti + i * 16 + half * 8 + r;
        const int col = tj + j * 16 + nlo;
        const size_t o = ob + (size_t)row * 256 + col;
        const float v = acc[i][j][r];
        if (mode == MODE_GRAM) {
          outF[o] = v * (1.f / (float)MPIX);
        } else if (mode == MODE_COPY) {
          outBf[o] = f2bf(v);
        } else if (mode == MODE_NS) {
          outBf[o] = f2bf(0.5f * ((row == col ? 3.f : 0.f) - v));
        } else { // MODE_FINAL
          outF[o] = v * sc;
        }
      }
}

// ---------------------------------------------------------------------------
// K3: trace of cov per batch -> 1/tr and sqrt(tr).  grid = NB, block = 256.
// ---------------------------------------------------------------------------
__global__ void trace_kernel(const float* __restrict__ cov,
                             float* __restrict__ trInv,
                             float* __restrict__ trSqrt) {
  const int b = blockIdx.x, t = threadIdx.x;
  __shared__ float red[256];
  red[t] = cov[(size_t)b * 65536 + (size_t)t * 257];   // diagonal
  __syncthreads();
  for (int s = 128; s > 0; s >>= 1) {
    if (t < s) red[t] += red[t + s];
    __syncthreads();
  }
  if (t == 0) {
    const float tr = red[0];
    trInv[b]  = 1.f / tr;
    trSqrt[b] = sqrtf(tr);
  }
}

// ---------------------------------------------------------------------------
// K4: A = cov/tr (bf16), Z0 = ZY0 = 0.5*(3I - A) (bf16).
// ---------------------------------------------------------------------------
__global__ void ns_init_kernel(const float* __restrict__ cov,
                               const float* __restrict__ trInv,
                               u16* __restrict__ Abf,
                               u16* __restrict__ Z0) {
  const size_t idx = (size_t)blockIdx.x * 256 + threadIdx.x; // < NB*65536
  const int b   = (int)(idx >> 16);
  const int rc  = (int)(idx & 65535);
  const int row = rc >> 8, col = rc & 255;
  const float a = cov[idx] * trInv[b];
  Abf[idx] = f2bf(a);
  Z0[idx]  = f2bf(0.5f * ((row == col ? 3.f : 0.f) - a));
}

// ---------------------------------------------------------------------------
// K6: v = column-mean(YZY); h = v@w1^T+b1; g = sigmoid(h@w2^T+b2);
// gp1 = 1+g. grid = NB, block = 256.
// ---------------------------------------------------------------------------
__global__ void se_gate_kernel(const float* __restrict__ YZY,
                               const float* __restrict__ w1,
                               const float* __restrict__ b1,
                               const float* __restrict__ w2,
                               const float* __restrict__ b2,
                               float* __restrict__ gp1) {
  const int b = blockIdx.x, t = threadIdx.x;
  __shared__ float v[256];
  __shared__ float h[96];
  const float* M = YZY + (size_t)b * 65536;

  float s = 0.f;
  for (int m = 0; m < 256; ++m) s += M[(size_t)m * 256 + t];
  v[t] = s * (1.f / 256.f);
  __syncthreads();

  if (t < CR) {
    float a = b1[t];
    const float* wr = w1 + (size_t)t * 256;
    for (int c = 0; c < 256; ++c) a += v[c] * wr[c];
    h[t] = a;
  }
  __syncthreads();

  float a = b2[t];
  const float* wr = w2 + (size_t)t * CR;
  for (int j = 0; j < CR; ++j) a += h[j] * wr[j];
  gp1[(size_t)b * 256 + t] = 1.f + 1.f / (1.f + expf(-a));
}

// ---------------------------------------------------------------------------
// K7: fused per-channel scale + bilinear upsample (align_corners=True).
// bilinear(s*x) = s*bilinear(x), so scale folds into the interpolation.
// ---------------------------------------------------------------------------
__global__ void upsample_scale_kernel(const float* __restrict__ x,
                                      const float* __restrict__ gp1,
                                      float* __restrict__ out,
                                      size_t total) {
  const size_t idx = (size_t)blockIdx.x * blockDim.x + threadIdx.x;
  if (idx >= total) return;
  const int ox = (int)(idx % OUTW);
  const size_t t1 = idx / OUTW;
  const int oy = (int)(t1 % OUTH);
  const size_t bc = t1 / OUTH;                 // < NB*NC

  const float ry = (float)oy * ((float)(IH - 1) / (float)(OUTH - 1));
  const float rx = (float)ox * ((float)(IW - 1) / (float)(OUTW - 1));
  const int y0 = (int)ry, x0 = (int)rx;
  const int y1 = min(y0 + 1, IH - 1);
  const int x1 = min(x0 + 1, IW - 1);
  const float wy = ry - (float)y0;
  const float wx = rx - (float)x0;

  const float* p = x + bc * (size_t)(IH * IW);
  const float v00 = p[y0 * IW + x0], v01 = p[y0 * IW + x1];
  const float v10 = p[y1 * IW + x0], v11 = p[y1 * IW + x1];
  const float top = v00 + (v01 - v00) * wx;
  const float bot = v10 + (v11 - v10) * wx;
  out[idx] = (top + (bot - top) * wy) * gp1[bc];
}

// ---------------------------------------------------------------------------
// Workspace layout (bytes, 256-aligned)
// ---------------------------------------------------------------------------
static constexpr size_t OFF_XC  = 0;                                   // bf16 16*256*1600
static constexpr size_t OFF_COV = OFF_XC + (size_t)NB * NC * KPAD * 2; // f32  16*256*256
static constexpr size_t OFF_TR  = OFF_COV + (size_t)NB * NC * NC * 4;  // 2*NB floats
static constexpr size_t MATB    = (size_t)NB * NC * NC * 2;            // one bf16 matrix set
static constexpr size_t OFF_ABF = OFF_TR + 256;
static constexpr size_t OFF_Y0  = OFF_ABF + MATB;
static constexpr size_t OFF_Y1  = OFF_Y0 + MATB;
static constexpr size_t OFF_Z0  = OFF_Y1 + MATB;
static constexpr size_t OFF_Z1  = OFF_Z0 + MATB;
static constexpr size_t OFF_ZY  = OFF_Z1 + MATB;
static constexpr size_t OFF_YZY = OFF_ZY + MATB;                       // f32
static constexpr size_t OFF_GP1 = OFF_YZY + (size_t)NB * NC * NC * 4;  // f32 NB*NC

extern "C" void kernel_launch(void* const* d_in, const int* in_sizes, int n_in,
                              void* d_out, int out_size, void* d_ws, size_t ws_size,
                              hipStream_t stream) {
  (void)in_sizes; (void)n_in; (void)out_size; (void)ws_size;
  const float* x  = (const float*)d_in[0];
  const float* w1 = (const float*)d_in[1];
  const float* b1 = (const float*)d_in[2];
  const float* w2 = (const float*)d_in[3];
  const float* b2 = (const float*)d_in[4];
  float* out = (float*)d_out;

  char* ws = (char*)d_ws;
  u16*   xc     = (u16*)(ws + OFF_XC);
  float* cov    = (float*)(ws + OFF_COV);
  float* trInv  = (float*)(ws + OFF_TR);
  float* trSqrt = (float*)(ws + OFF_TR + 64);
  u16*   Abf    = (u16*)(ws + OFF_ABF);
  u16*   Ybuf[2] = { (u16*)(ws + OFF_Y0), (u16*)(ws + OFF_Y1) };
  u16*   Zbuf[2] = { (u16*)(ws + OFF_Z0), (u16*)(ws + OFF_Z1) };
  u16*   ZY     = (u16*)(ws + OFF_ZY);
  float* YZY    = (float*)(ws + OFF_YZY);
  float* gp1    = (float*)(ws + OFF_GP1);

  const dim3 mmGrid(8, 8, NB);
  const dim3 mmBlk(32, 1, 1);

  // 1) pool + center + bf16
  pool_center_kernel<<<NB * NC, 256, 0, stream>>>(x, xc);

  // 2) cov = (1/M) Xc Xc^T   (bf16 WMMA Gram, K padded to 1600 with zeros)
  mm256_bf16_kernel<<<mmGrid, mmBlk, 0, stream>>>(
      xc, xc, KPAD, KPAD, KPAD, nullptr, cov, MODE_GRAM, nullptr);

  // 3) trace -> 1/tr, sqrt(tr)
  trace_kernel<<<NB, 256, 0, stream>>>(cov, trInv, trSqrt);

  // 4) A = cov/tr ; Z0 = ZY0 = 0.5*(3I - A)
  ns_init_kernel<<<NB * 256, 256, 0, stream>>>(cov, trInv, Abf, Zbuf[0]);

  // 5) Y0 = A @ ZY0
  mm256_bf16_kernel<<<mmGrid, mmBlk, 0, stream>>>(
      Abf, Zbuf[0], 256, 256, 256, Ybuf[0], nullptr, MODE_COPY, nullptr);

  // 6) Newton-Schulz iterations (all operands symmetric: polynomials in A)
  int cur = 0;
  for (int it = 0; it < 3; ++it) {
    const int nxt = cur ^ 1;
    mm256_bf16_kernel<<<mmGrid, mmBlk, 0, stream>>>(          // ZY = 0.5*(3I - Z@Y)
        Zbuf[cur], Ybuf[cur], 256, 256, 256, ZY, nullptr, MODE_NS, nullptr);
    mm256_bf16_kernel<<<mmGrid, mmBlk, 0, stream>>>(          // Y' = Y@ZY
        Ybuf[cur], ZY, 256, 256, 256, Ybuf[nxt], nullptr, MODE_COPY, nullptr);
    mm256_bf16_kernel<<<mmGrid, mmBlk, 0, stream>>>(          // Z' = ZY@Z
        ZY, Zbuf[cur], 256, 256, 256, Zbuf[nxt], nullptr, MODE_COPY, nullptr);
    cur = nxt;
  }

  // 7) final: ZY = 0.5*(3I - Z@Y); YZY = (Y@ZY) * sqrt(tr)
  mm256_bf16_kernel<<<mmGrid, mmBlk, 0, stream>>>(
      Zbuf[cur], Ybuf[cur], 256, 256, 256, ZY, nullptr, MODE_NS, nullptr);
  mm256_bf16_kernel<<<mmGrid, mmBlk, 0, stream>>>(
      Ybuf[cur], ZY, 256, 256, 256, nullptr, YZY, MODE_FINAL, trSqrt);

  // 8) gate: column mean -> FC(256->85) -> FC(85->256) -> sigmoid -> 1+g
  se_gate_kernel<<<NB, 256, 0, stream>>>(YZY, w1, b1, w2, b2, gp1);

  // 9) fused per-channel scale + bilinear upsample
  const size_t total = (size_t)NB * NC * OUTH * OUTW;    // 103,342,080
  const unsigned nblk = (unsigned)((total + 255) / 256);
  upsample_scale_kernel<<<nblk, 256, 0, stream>>>(x, gp1, out, total);
}
